// Attention_53060025975012
// MI455X (gfx1250) — matrix operand order repaired and checked by reading, not compiled, not run on hardware
//
#include <hip/hip_runtime.h>
#include <stdint.h>

// ---------------------------------------------------------------------------
// Shapes: x(4,64,256,384) f32 -> out(4,64,256,384) f32. H=6, D=64, BN=256.
// All matmuls on v_wmma_f32_16x16x32_bf16 (fp32 accumulate). Attention stages
// K and V^T in LDS via global_load_async_to_lds_b128 (ASYNCcnt path).
// ---------------------------------------------------------------------------

typedef __attribute__((ext_vector_type(16))) __bf16 v16bf;
typedef __attribute__((ext_vector_type(8)))  __bf16 v8bf;
typedef __attribute__((ext_vector_type(8)))  float  v8f;

union ABFrag { v16bf v; v8bf h[2]; };

// A fragment (16x32 bf16, MxK): lane m = L%16; elems 0..7 = K[kt*32+half*8 ..],
// elems 8..15 = K[kt*32+16+half*8 ..]   (ISA 7.12.2 16-bit A table)
static __device__ __forceinline__ v16bf load_a_frag(const __bf16* row, int kt, int half) {
  ABFrag f;
  const __bf16* p = row + kt * 32 + half * 8;
  f.h[0] = *(const v8bf*)(p);
  f.h[1] = *(const v8bf*)(p + 16);
  return f.v;
}
// B fragment (32x16 bf16, KxN): lane n = L%16; elems 0..7 = K[kt*32+half*8 ..],
// elems 8..15 = K[kt*32+16+half*8 ..]   (same k order as A)
static __device__ __forceinline__ v16bf load_b_frag(const __bf16* row, int kt, int half) {
  ABFrag f;
  const __bf16* p = row + kt * 32 + half * 8;
  f.h[0] = *(const v8bf*)(p);
  f.h[1] = *(const v8bf*)(p + 16);
  return f.v;
}

#define WMMA_BF16(a, b, c) \
  __builtin_amdgcn_wmma_f32_16x16x32_bf16(false, (a), false, (b), (short)0, (c), false, false)

// ---------------------------------------------------------------------- utils
__global__ void cvt_f32_to_bf16(const float* __restrict__ in, __bf16* __restrict__ out, int n) {
  int i = blockIdx.x * blockDim.x + threadIdx.x;
  if (i < n) out[i] = (__bf16)in[i];
}

// in[K][N] f32 -> out[N][K] bf16 (GEMM B fragments then read contiguous K)
__global__ void cvt_transpose_bf16(const float* __restrict__ in, __bf16* __restrict__ out,
                                   int K, int N) {
  int i = blockIdx.x * blockDim.x + threadIdx.x;
  if (i < K * N) {
    int k = i / N, n = i % N;
    out[(size_t)n * K + k] = (__bf16)in[i];
  }
}

// ------------------------------------------------------------------ QKV GEMM
// X[65536][384] bf16 @ WqkvT -> q,k row-major [bnh][p][64], v transposed [bnh][64][p]
// grid: (512 m-blocks of 128 rows, 18 n-blocks of 64 cols), 256 thr = 8 waves.
// Fully-unrolled, register double-buffered K-loop: fragments for kt+1 are in
// flight while the WMMAs for kt execute (SSA renaming removes buffer copies).
__global__ __launch_bounds__(256) void qkv_gemm(const __bf16* __restrict__ X,
                                                const __bf16* __restrict__ WT,
                                                __bf16* __restrict__ Q,
                                                __bf16* __restrict__ Kd,
                                                __bf16* __restrict__ VT) {
  const int wave = threadIdx.x >> 5, lane = threadIdx.x & 31;
  const int half = lane >> 4, lan16 = lane & 15;
  const int row_base = blockIdx.x * 128 + wave * 16;
  const int nb = blockIdx.y;                       // 64-wide column block (0..17)
  const int n_base = nb * 64;

  const __bf16* arow = X + (size_t)(row_base + lan16) * 384;
  const __bf16* brow[4];
#pragma unroll
  for (int t = 0; t < 4; ++t) brow[t] = WT + (size_t)(n_base + t * 16 + lan16) * 384;

  v8f acc[4] = {};
  v16bf a_cur = load_a_frag(arow, 0, half);
  v16bf b_cur[4];
#pragma unroll
  for (int t = 0; t < 4; ++t) b_cur[t] = load_b_frag(brow[t], 0, half);

#pragma unroll
  for (int kt = 0; kt < 12; ++kt) {
    v16bf a_nxt, b_nxt[4];
    if (kt < 11) {                                 // issue next loads first
      a_nxt = load_a_frag(arow, kt + 1, half);
#pragma unroll
      for (int t = 0; t < 4; ++t) b_nxt[t] = load_b_frag(brow[t], kt + 1, half);
    }
#pragma unroll
    for (int t = 0; t < 4; ++t) acc[t] = WMMA_BF16(a_cur, b_cur[t], acc[t]);
    if (kt < 11) {
      a_cur = a_nxt;
#pragma unroll
      for (int t = 0; t < 4; ++t) b_cur[t] = b_nxt[t];
    }
  }

  // column block nb -> (s = q/k/v, head h): col = s*384 + h*64 + d
  const int s = nb / 6, h = nb % 6;
  const int bn = row_base >> 8;
  const int p_base = row_base & 255;
  const size_t bnh = (size_t)bn * 6 + h;
#pragma unroll
  for (int t = 0; t < 4; ++t) {
    const int d = t * 16 + lan16;
#pragma unroll
    for (int r = 0; r < 8; ++r) {
      const int p = p_base + half * 8 + r;         // C layout: VGPR r -> row r + 8*half
      const __bf16 bv = (__bf16)acc[t][r];
      if (s == 0)      Q [(bnh * 256 + p) * 64 + d] = bv;
      else if (s == 1) Kd[(bnh * 256 + p) * 64 + d] = bv;
      else             VT[(bnh * 64 + d) * 256 + p] = bv;   // v stored transposed
    }
  }
}

// ------------------------------------------------------------------ attention
// One block per (b,n,h). K (256x64) and V^T (64x256) are async-copied to LDS
// once per block (ASYNCcnt), then every wave's B fragments are ds_load_b128s.
// S = q@k^T via WMMA, *64 scale, in-register softmax (shfl within 16-lane
// halves), P restaged through LDS (C layout -> A layout), O = P@v via WMMA.
__global__ __launch_bounds__(256) void attn_kernel(const __bf16* __restrict__ Q,
                                                   const __bf16* __restrict__ Kd,
                                                   const __bf16* __restrict__ VT,
                                                   __bf16* __restrict__ O) {
  __shared__ __align__(64) __bf16 Ksh[256 * 64];      // 32 KB
  __shared__ __align__(64) __bf16 Vsh[64 * 256];      // 32 KB
  __shared__ __align__(64) __bf16 Pst[8][16 * 256];   // 64 KB per-wave P staging
  const int tid = threadIdx.x;
  const int wave = tid >> 5, lane = tid & 31;
  const int half = lane >> 4, lan16 = lane & 15;
  const int bnh = blockIdx.x;                         // 0..1535
  const int bn = bnh / 6, h = bnh % 6;
  const __bf16* qbase = Q + (size_t)bnh * 256 * 64;
  const char* kg = (const char*)(Kd + (size_t)bnh * 256 * 64);
  const char* vg = (const char*)(VT + (size_t)bnh * 64 * 256);

  // ---- async stage K and V^T into LDS: per-lane b128s, lane-major so each
  // wave moves contiguous 512B spans on both the global and LDS side.
  {
    const unsigned kl = (unsigned)(uintptr_t)&Ksh[0];
    const unsigned vl = (unsigned)(uintptr_t)&Vsh[0];
#pragma unroll
    for (int i = 0; i < 8; ++i) {
      const unsigned boff = (unsigned)(i * 256 + tid) * 16;   // 8*256*16 = 32 KB
      const uint64_t gk = (uint64_t)(uintptr_t)(kg + boff);
      const uint64_t gv = (uint64_t)(uintptr_t)(vg + boff);
      asm volatile("global_load_async_to_lds_b128 %0, %1, off"
                   :: "v"(kl + boff), "v"(gk) : "memory");
      asm volatile("global_load_async_to_lds_b128 %0, %1, off"
                   :: "v"(vl + boff), "v"(gv) : "memory");
    }
    asm volatile("s_wait_asynccnt 0" ::: "memory");
  }
  __syncthreads();

  __bf16* Pw = &Pst[wave][0];

  for (int rb = wave; rb < 16; rb += 8) {             // two 16-row strips per wave
    const int p_base = rb * 16;
    const __bf16* qrow = qbase + (size_t)(p_base + lan16) * 64;
    const v16bf qa0 = load_a_frag(qrow, 0, half);
    const v16bf qa1 = load_a_frag(qrow, 1, half);

    v8f S[16];
#pragma unroll
    for (int t = 0; t < 16; ++t) {
      const __bf16* krow = Ksh + (t * 16 + lan16) * 64;
      v8f a = {};
      a = WMMA_BF16(qa0, load_b_frag(krow, 0, half), a);
      a = WMMA_BF16(qa1, load_b_frag(krow, 1, half), a);
      S[t] = a;
    }

    // softmax rows (reference scale: multiply scores by D = 64)
    float rmax[8], rinv[8];
#pragma unroll
    for (int r = 0; r < 8; ++r) {
      float mv = S[0][r];
#pragma unroll
      for (int t = 1; t < 16; ++t) mv = fmaxf(mv, S[t][r]);
      for (int m = 1; m < 16; m <<= 1) mv = fmaxf(mv, __shfl_xor(mv, m, 32));
      rmax[r] = mv;
    }
#pragma unroll
    for (int r = 0; r < 8; ++r) {
      float sum = 0.f;
#pragma unroll
      for (int t = 0; t < 16; ++t) {
        float e = __expf(64.0f * (S[t][r] - rmax[r]));
        S[t][r] = e;
        sum += e;
      }
      for (int m = 1; m < 16; m <<= 1) sum += __shfl_xor(sum, m, 32);
      rinv[r] = 1.0f / sum;
    }

    // C layout -> row-major P[16][256] in LDS so PV A-fragments are contiguous
#pragma unroll
    for (int t = 0; t < 16; ++t)
#pragma unroll
      for (int r = 0; r < 8; ++r)
        Pw[(half * 8 + r) * 256 + (t * 16 + lan16)] = (__bf16)(S[t][r] * rinv[r]);

    asm volatile("s_wait_dscnt 0" ::: "memory");      // P visible before ds_load_b128s

    v8f Oacc[4] = {};
    const __bf16* prow = Pw + lan16 * 256;
#pragma unroll
    for (int kt = 0; kt < 8; ++kt) {                  // K = 256 over softmax axis
      v16bf pa = load_a_frag(prow, kt, half);
#pragma unroll
      for (int t = 0; t < 4; ++t) {
        const __bf16* vrow = Vsh + (t * 16 + lan16) * 256;
        Oacc[t] = WMMA_BF16(pa, load_b_frag(vrow, kt, half), Oacc[t]);
      }
    }

    // store token-major [bn*256+p][h*64+d] for the proj GEMM
#pragma unroll
    for (int t = 0; t < 4; ++t) {
      const int d = t * 16 + lan16;
#pragma unroll
      for (int r = 0; r < 8; ++r) {
        const int p = p_base + half * 8 + r;
        O[((size_t)bn * 256 + p) * 384 + h * 64 + d] = (__bf16)Oacc[t][r];
      }
    }
  }
}

// ------------------------------------------------------------------ Proj GEMM
// O[65536][384] bf16 @ WprojT + bias -> out f32. grid (512, 6), 256 thr.
__global__ __launch_bounds__(256) void proj_gemm(const __bf16* __restrict__ A,
                                                 const __bf16* __restrict__ WT,
                                                 const float* __restrict__ bias,
                                                 float* __restrict__ out) {
  const int wave = threadIdx.x >> 5, lane = threadIdx.x & 31;
  const int half = lane >> 4, lan16 = lane & 15;
  const size_t row_base = (size_t)blockIdx.x * 128 + wave * 16;
  const int n_base = blockIdx.y * 64;

  const __bf16* arow = A + (row_base + lan16) * 384;
  const __bf16* brow[4];
#pragma unroll
  for (int t = 0; t < 4; ++t) brow[t] = WT + (size_t)(n_base + t * 16 + lan16) * 384;

  v8f acc[4] = {};
  v16bf a_cur = load_a_frag(arow, 0, half);
  v16bf b_cur[4];
#pragma unroll
  for (int t = 0; t < 4; ++t) b_cur[t] = load_b_frag(brow[t], 0, half);

#pragma unroll
  for (int kt = 0; kt < 12; ++kt) {
    v16bf a_nxt, b_nxt[4];
    if (kt < 11) {                                 // issue next loads first
      a_nxt = load_a_frag(arow, kt + 1, half);
#pragma unroll
      for (int t = 0; t < 4; ++t) b_nxt[t] = load_b_frag(brow[t], kt + 1, half);
    }
#pragma unroll
    for (int t = 0; t < 4; ++t) acc[t] = WMMA_BF16(a_cur, b_cur[t], acc[t]);
    if (kt < 11) {
      a_cur = a_nxt;
#pragma unroll
      for (int t = 0; t < 4; ++t) b_cur[t] = b_nxt[t];
    }
  }

#pragma unroll
  for (int t = 0; t < 4; ++t) {
    const int n = n_base + t * 16 + lan16;
    const float bv = bias[n];
#pragma unroll
    for (int r = 0; r < 8; ++r) {
      const size_t row = row_base + half * 8 + r;
      out[row * 384 + n] = acc[t][r] + bv;   // 16 consecutive f32 per row: coalesced
    }
  }
}

// --------------------------------------------------------------------- launch
extern "C" void kernel_launch(void* const* d_in, const int* in_sizes, int n_in,
                              void* d_out, int out_size, void* d_ws, size_t ws_size,
                              hipStream_t stream) {
  const float* x      = (const float*)d_in[0];   // (4,64,256,384)
  const float* w_qkv  = (const float*)d_in[1];   // (384,1152)
  const float* w_proj = (const float*)d_in[2];   // (384,384)
  const float* b_proj = (const float*)d_in[3];   // (384,)
  float* out = (float*)d_out;

  const int TOK = 4 * 64 * 256;                  // 65536 tokens
  const int C = 384, C3 = 1152;

  // workspace carve-up (offsets 256B-aligned); x_bf16 region reused for O
  char* ws = (char*)d_ws;
  const size_t Sx = (size_t)TOK * C * 2;         // 50331648
  __bf16* x_bf = (__bf16*)(ws);
  __bf16* wqT  = (__bf16*)(ws + Sx);
  __bf16* wpT  = (__bf16*)(ws + Sx + 884736);
  __bf16* Qb   = (__bf16*)(ws + Sx + 884736 + 294912);
  __bf16* Kb   = (__bf16*)((char*)Qb + Sx);
  __bf16* VTb  = (__bf16*)((char*)Kb + Sx);
  __bf16* Ob   = x_bf;                           // dead after qkv_gemm

  // 1) precision conversion / weight transposes
  cvt_f32_to_bf16<<<(TOK * C + 255) / 256, 256, 0, stream>>>(x, x_bf, TOK * C);
  cvt_transpose_bf16<<<(C * C3 + 255) / 256, 256, 0, stream>>>(w_qkv, wqT, C, C3);
  cvt_transpose_bf16<<<(C * C + 255) / 256, 256, 0, stream>>>(w_proj, wpT, C, C);

  // 2) fused QKV projection (65536 x 1152 x 384)
  qkv_gemm<<<dim3(TOK / 128, C3 / 64), 256, 0, stream>>>(x_bf, wqT, Qb, Kb, VTb);

  // 3) attention per (b,n,h): 1536 x (256x256x64 + softmax + 256x64x256)
  attn_kernel<<<1536, 256, 0, stream>>>(Qb, Kb, VTb, Ob);

  // 4) output projection + bias (65536 x 384 x 384)
  proj_gemm<<<dim3(TOK / 128, C / 64), 256, 0, stream>>>(Ob, wpT, b_proj, out);
}
